// Attention_49160195670039
// MI455X (gfx1250) — matrix operand order repaired and checked by reading, not compiled, not run on hardware
//
#include <hip/hip_runtime.h>

#define B_ 64
#define L_ 1024
#define Q_ 128
#define D_ 256

// ---- CDNA5 WMMA types -------------------------------------------------------
typedef __attribute__((ext_vector_type(16))) __bf16 v16bf;
typedef __attribute__((ext_vector_type(8)))  float  v8f;

union FragBF { uint4 u4[2]; v16bf v; };

__device__ __forceinline__ unsigned short f2bf(float f) {
  union { float f; unsigned u; } c{f};
  unsigned r = c.u + 0x7FFFu + ((c.u >> 16) & 1u);   // round-to-nearest-even
  return (unsigned short)(r >> 16);
}

// LDS layout (bf16 element units), pitches padded for bank-conflict-free b128:
//   qlds : 128 x 264   query, row-major (q major)        -> B of sim GEMM
//   qtlds: 256 x 136   query, transposed (d major)       -> B of a   GEMM
//   clds : 128 x 264   context*w_cq tile, row-major      -> A of sim GEMM
//   alds : 128 x 136   alpha tile (per-wave scratch)     -> A of a   GEMM
#define QLDS_OFF  0
#define QTLDS_OFF (128 * 264)
#define CLDS_OFF  (QTLDS_OFF + 256 * 136)
#define ALDS_OFF  (CLDS_OFF + 128 * 264)
#define FLT_OFF_B ((ALDS_OFF + 128 * 136) * 2)          // byte offset of float region
#define SMEM_BYTES (FLT_OFF_B + (768 + 128 + 128) * 4)  // + w(768) + s_q(128) + s_c(128)

__global__ void __launch_bounds__(256)
attn_main(const float* __restrict__ context,
          const float* __restrict__ cmask,
          const float* __restrict__ query,
          const float* __restrict__ qmask,
          const float* __restrict__ attn_w,
          const float* __restrict__ attn_b,
          float* __restrict__ out,
          float* __restrict__ q2c_ws)
{
  extern __shared__ char smem[];
  unsigned short* qlds  = (unsigned short*)smem + QLDS_OFF;
  unsigned short* qtlds = (unsigned short*)smem + QTLDS_OFF;
  unsigned short* clds  = (unsigned short*)smem + CLDS_OFF;
  unsigned short* alds  = (unsigned short*)smem + ALDS_OFF;
  float* wlds  = (float*)(smem + FLT_OFF_B);
  float* sqlds = wlds + 768;
  float* sclds = sqlds + 128;

  const int tid  = threadIdx.x;
  const int wave = tid >> 5;
  const int lane = tid & 31;
  const int n    = lane & 15;      // column-within-tile (N) for WMMA C layout
  const int hs   = lane >> 4;      // half-select (rows 0..7 vs 8..15)
  const int b    = blockIdx.x >> 3;
  const int l0   = (blockIdx.x & 7) * 128 + wave * 16;

  // ---- phase 0: weights to LDS ---------------------------------------------
  for (int i = tid; i < 768; i += 256) wlds[i] = attn_w[i];
  __syncthreads();

  // ---- phase 1: query -> qlds (bf16 row-major) + qtlds (bf16 d-major) + s_q
  {
    const int q  = tid >> 1;
    const int db = (tid & 1) * 128;
    const float4* src = (const float4*)(query + ((size_t)b * Q_ + q) * D_ + db);
    float acc = 0.f;
    #pragma unroll 4
    for (int i = 0; i < 32; ++i) {
      float4 v = src[i];
      int d = db + i * 4;
      unsigned short h0 = f2bf(v.x), h1 = f2bf(v.y), h2 = f2bf(v.z), h3 = f2bf(v.w);
      *(unsigned int*)&qlds[q * 264 + d]     = (unsigned)h0 | ((unsigned)h1 << 16);
      *(unsigned int*)&qlds[q * 264 + d + 2] = (unsigned)h2 | ((unsigned)h3 << 16);
      qtlds[(d    ) * 136 + q] = h0;
      qtlds[(d + 1) * 136 + q] = h1;
      qtlds[(d + 2) * 136 + q] = h2;
      qtlds[(d + 3) * 136 + q] = h3;
      acc += v.x * wlds[256 + d]     + v.y * wlds[256 + d + 1]
           + v.z * wlds[256 + d + 2] + v.w * wlds[256 + d + 3];
    }
    acc += __shfl_xor(acc, 1, 32);
    if ((tid & 1) == 0) sqlds[q] = acc;
  }

  // ---- phase 2: context tile * w_cq -> clds (bf16) + s_c -------------------
  {
    const int db = hs * 128;
    const float4* src = (const float4*)(context + ((size_t)b * L_ + l0 + n) * D_ + db);
    unsigned short* crow = &clds[(wave * 16 + n) * 264];
    float acc = 0.f;
    #pragma unroll 4
    for (int i = 0; i < 32; ++i) {
      float4 v = src[i];
      int d = db + i * 4;
      unsigned short h0 = f2bf(v.x * wlds[512 + d]);
      unsigned short h1 = f2bf(v.y * wlds[512 + d + 1]);
      unsigned short h2 = f2bf(v.z * wlds[512 + d + 2]);
      unsigned short h3 = f2bf(v.w * wlds[512 + d + 3]);
      *(unsigned int*)&crow[d]     = (unsigned)h0 | ((unsigned)h1 << 16);
      *(unsigned int*)&crow[d + 2] = (unsigned)h2 | ((unsigned)h3 << 16);
      acc += v.x * wlds[d]     + v.y * wlds[d + 1]
           + v.z * wlds[d + 2] + v.w * wlds[d + 3];
    }
    acc += __shfl_xor(acc, 16, 32);
    if (hs == 0) sclds[wave * 16 + n] = acc;
  }
  __syncthreads();

  // ---- phase 3: sim tile (16 x 128) via v_wmma_f32_16x16x32_bf16 -----------
  const float bias = attn_b[0];
  v8f simacc[8];
  #pragma unroll
  for (int qt = 0; qt < 8; ++qt) {
    float cadd = sqlds[qt * 16 + n] + bias;      // s_q + b (per column)
    #pragma unroll
    for (int j = 0; j < 8; ++j)
      simacc[qt][j] = cadd + sclds[wave * 16 + hs * 8 + j];   // + s_c (per row)
  }

  #pragma unroll
  for (int kb = 0; kb < 8; ++kb) {               // K = 256, 32 per step
    FragBF A;
    const unsigned short* ar = &clds[(wave * 16 + n) * 264 + kb * 32 + hs * 8];
    A.u4[0] = *(const uint4*)ar;                 // K {0..7} / {8..15}
    A.u4[1] = *(const uint4*)(ar + 16);          // K {16..23} / {24..31}
    #pragma unroll
    for (int qt = 0; qt < 8; ++qt) {
      FragBF Bf;
      const unsigned short* br = &qlds[(qt * 16 + n) * 264 + kb * 32 + hs * 8];
      Bf.u4[0] = *(const uint4*)br;              // K {0..7} / {8..15}
      Bf.u4[1] = *(const uint4*)(br + 16);       // K {16..23} / {24..31}
      simacc[qt] = __builtin_amdgcn_wmma_f32_16x16x32_bf16(
          false, A.v, false, Bf.v, (short)0, simacc[qt], false, false);
    }
  }

  // ---- phase 4: softmax over q, q2c row-max, alpha(bf16) -> alds -----------
  float qmv[8];
  #pragma unroll
  for (int qt = 0; qt < 8; ++qt) qmv[qt] = qmask[b * Q_ + qt * 16 + n];

  #pragma unroll
  for (int j = 0; j < 8; ++j) {
    const int l = l0 + hs * 8 + j;
    float mx = simacc[0][j];
    #pragma unroll
    for (int qt = 1; qt < 8; ++qt) mx = fmaxf(mx, simacc[qt][j]);
    #pragma unroll
    for (int s = 8; s >= 1; s >>= 1) mx = fmaxf(mx, __shfl_xor(mx, s, 32));
    if (n == 0) q2c_ws[(size_t)b * L_ + l] = mx;
    float sum = 0.f;
    #pragma unroll
    for (int qt = 0; qt < 8; ++qt) {
      float e = __expf(simacc[qt][j] - mx);
      simacc[qt][j] = e;
      sum += e;
    }
    #pragma unroll
    for (int s = 8; s >= 1; s >>= 1) sum += __shfl_xor(sum, s, 32);
    float scale = cmask[b * L_ + l] / sum;       // masks applied after softmax
    #pragma unroll
    for (int qt = 0; qt < 8; ++qt)
      alds[(wave * 16 + hs * 8 + j) * 136 + qt * 16 + n] =
          f2bf(simacc[qt][j] * scale * qmv[qt]);
  }
  asm volatile("s_wait_dscnt 0" ::: "memory");   // alpha LDS transpose complete (wave-local)

  // ---- phase 5: a = alpha @ query  (16 x 256) ------------------------------
  v8f aacc[16];
  const v8f vzero = {0.f, 0.f, 0.f, 0.f, 0.f, 0.f, 0.f, 0.f};
  #pragma unroll
  for (int dt = 0; dt < 16; ++dt) aacc[dt] = vzero;

  #pragma unroll
  for (int kb = 0; kb < 4; ++kb) {               // K = 128, 32 per step
    FragBF A;
    const unsigned short* ar = &alds[(wave * 16 + n) * 136 + kb * 32 + hs * 8];
    A.u4[0] = *(const uint4*)ar;
    A.u4[1] = *(const uint4*)(ar + 16);
    #pragma unroll
    for (int dt = 0; dt < 16; ++dt) {
      FragBF Bf;
      const unsigned short* br = &qtlds[(dt * 16 + n) * 136 + kb * 32 + hs * 8];
      Bf.u4[0] = *(const uint4*)br;
      Bf.u4[1] = *(const uint4*)(br + 16);
      aacc[dt] = __builtin_amdgcn_wmma_f32_16x16x32_bf16(
          false, A.v, false, Bf.v, (short)0, aacc[dt], false, false);
    }
  }

  // ---- phase 6: write output slices 1 (a) and 2 (context*a) ---------------
  #pragma unroll 2
  for (int dt = 0; dt < 16; ++dt) {
    const int d = dt * 16 + n;
    #pragma unroll
    for (int j = 0; j < 8; ++j) {
      const size_t row = (size_t)b * L_ + (l0 + hs * 8 + j);
      float av = aacc[dt][j];
      float cv = context[row * D_ + d];
      out[row * (4 * D_) + D_ + d]     = av;
      out[row * (4 * D_) + 2 * D_ + d] = cv * av;
    }
  }
}

// ---- kernel 2: beta = softmax_l(q2c)*cmask ; b = beta^T . context ----------
__global__ void __launch_bounds__(256)
attn_beta(const float* __restrict__ context,
          const float* __restrict__ cmask,
          const float* __restrict__ q2c_ws,
          float* __restrict__ bvec_ws)
{
  __shared__ float betas[L_];
  __shared__ float red[256];
  const int tid = threadIdx.x;
  const int b   = blockIdx.x;

  float v[4];
  float mx = -3.4e38f;
  #pragma unroll
  for (int k = 0; k < 4; ++k) {
    v[k] = q2c_ws[(size_t)b * L_ + k * 256 + tid];
    mx = fmaxf(mx, v[k]);
  }
  red[tid] = mx; __syncthreads();
  for (int s = 128; s > 0; s >>= 1) {
    if (tid < s) red[tid] = fmaxf(red[tid], red[tid + s]);
    __syncthreads();
  }
  mx = red[0]; __syncthreads();

  float sum = 0.f;
  #pragma unroll
  for (int k = 0; k < 4; ++k) { v[k] = __expf(v[k] - mx); sum += v[k]; }
  red[tid] = sum; __syncthreads();
  for (int s = 128; s > 0; s >>= 1) {
    if (tid < s) red[tid] += red[tid + s];
    __syncthreads();
  }
  float inv = 1.0f / red[0];

  #pragma unroll
  for (int k = 0; k < 4; ++k) {
    int l = k * 256 + tid;
    betas[l] = v[k] * inv * cmask[(size_t)b * L_ + l];
  }
  __syncthreads();

  float acc = 0.f;
  const float* cb = context + (size_t)b * L_ * D_ + tid;   // d = tid, coalesced
  for (int l = 0; l < L_; ++l)
    acc += betas[l] * cb[(size_t)l * D_];
  bvec_ws[b * D_ + tid] = acc;
}

// ---- kernel 3: write output slices 0 (context) and 3 (context*b) -----------
__global__ void __launch_bounds__(256)
attn_out03(const float* __restrict__ context,
           const float* __restrict__ bvec_ws,
           float* __restrict__ out)
{
  size_t idx = (size_t)blockIdx.x * 256 + threadIdx.x;   // over B*L*D
  int d      = (int)(idx & (D_ - 1));
  size_t row = idx >> 8;                                  // b*L + l
  int b      = (int)(row >> 10);
  float cv = context[idx];
  float bv = bvec_ws[b * D_ + d];
  out[row * (4 * D_) + d]          = cv;
  out[row * (4 * D_) + 3 * D_ + d] = cv * bv;
}

extern "C" void kernel_launch(void* const* d_in, const int* in_sizes, int n_in,
                              void* d_out, int out_size, void* d_ws, size_t ws_size,
                              hipStream_t stream) {
  (void)in_sizes; (void)n_in; (void)out_size; (void)ws_size;
  const float* context = (const float*)d_in[0];
  const float* cmask   = (const float*)d_in[1];
  const float* query   = (const float*)d_in[2];
  const float* qmask   = (const float*)d_in[3];
  const float* attn_w  = (const float*)d_in[4];
  const float* attn_b  = (const float*)d_in[5];
  float* out  = (float*)d_out;
  float* q2c  = (float*)d_ws;          // B*L floats
  float* bvec = q2c + B_ * L_;         // B*D floats

  attn_main<<<dim3(B_ * (L_ / 128)), dim3(256), SMEM_BYTES, stream>>>(
      context, cmask, query, qmask, attn_w, attn_b, out, q2c);
  attn_beta<<<dim3(B_), dim3(256), 0, stream>>>(context, cmask, q2c, bvec);
  attn_out03<<<dim3((B_ * L_ * D_) / 256), dim3(256), 0, stream>>>(
      context, bvec, out);
}
